// NaiveAttentionModule_34376918237970
// MI455X (gfx1250) — matrix operand order repaired and checked by reading, not compiled, not run on hardware
//
#include <hip/hip_runtime.h>

// ---------------------------------------------------------------------------
// MI455X (gfx1250) causal MHA: B=2, S=2048, E=1024, H=16, D=64.
// All matmuls via v_wmma_f32_16x16x32_bf16 (f32 accumulate).
// Flash-attention keeps the 0.5GB (f32) score matrix out of memory entirely.
// LDS tiles filled with GLOBAL_LOAD_ASYNC_TO_LDS_B128 (ASYNCcnt) when the
// toolchain exposes the builtin; plain copies otherwise.
// ---------------------------------------------------------------------------

typedef __attribute__((ext_vector_type(16))) __bf16 v16bf;
typedef __attribute__((ext_vector_type(8)))  float  v8f;
typedef __attribute__((ext_vector_type(4)))  int    v4i;

union Frag16 {            // one lane's 16 bf16 (8 VGPRs) of an A/B fragment
    v16bf v;
    unsigned short u[16];
    uint4 q[2];
};
union FragF {             // one lane's 8 f32 (8 VGPRs) of a C/D fragment
    v8f v;
    float f[8];
};

__device__ __forceinline__ unsigned short f2bf(float x) {
    unsigned int u = __float_as_uint(x);
    unsigned int r = u + 0x7FFFu + ((u >> 16) & 1u);   // RNE
    return (unsigned short)(r >> 16);
}

__device__ __forceinline__ v8f wmma_bf16(v16bf a, v16bf b, v8f c) {
    return __builtin_amdgcn_wmma_f32_16x16x32_bf16(
        false, a, false, b, (short)0, c, false, false);
}

// ---------------------------------------------------------------------------
// Async global->LDS 16B copy (per lane), CDNA5 GLOBAL_LOAD_ASYNC_TO_LDS_B128.
// Builtin signature (from hipcc diagnostic): param0 = v4i in addrspace(1).
// ---------------------------------------------------------------------------
#if defined(__has_builtin)
#if __has_builtin(__builtin_amdgcn_global_load_async_to_lds_b128)
#define ASYNC_G2L 1
#endif
#endif

typedef __attribute__((address_space(1))) v4i gas_v4i;
typedef __attribute__((address_space(3))) v4i las_v4i;

__device__ __forceinline__ void g2l_b128(void* lds, const void* g) {
#ifdef ASYNC_G2L
    __builtin_amdgcn_global_load_async_to_lds_b128(
        (gas_v4i*)g, (las_v4i*)lds, 0, 0);
#else
    *(uint4*)lds = *(const uint4*)g;
#endif
}

__device__ __forceinline__ void async_join() {
#ifdef ASYNC_G2L
#if __has_builtin(__builtin_amdgcn_s_wait_asynccnt)
    __builtin_amdgcn_s_wait_asynccnt(0);
#else
    asm volatile("s_wait_asynccnt 0x0" ::: "memory");
#endif
#endif
}

// A-fragment (16x32, ISA 7.12.2): lane half h reads k chunks [8h..8h+7] and
// [16+8h..16+8h+7] from its row (row = lane&15).
__device__ __forceinline__ v16bf load_fragA(const unsigned short* row, int half) {
    Frag16 f;
    f.q[0] = *(const uint4*)(row + half * 8);
    f.q[1] = *(const uint4*)(row + 16 + half * 8);
    return f.v;
}
// B-fragment (32x16): lane half h reads k chunks [8h..8h+7] and
// [16+8h..16+8h+7] from the row of B^T (row = n = lane&15).
__device__ __forceinline__ v16bf load_fragB(const unsigned short* rowT, int half) {
    Frag16 f;
    f.q[0] = *(const uint4*)(rowT + half * 8);
    f.q[1] = *(const uint4*)(rowT + 16 + half * 8);
    return f.v;
}

// ---------------------------------------------------------------------------
// fp32 -> bf16 conversion
// ---------------------------------------------------------------------------
__global__ void cvt_f32_bf16(const float* __restrict__ in,
                             unsigned short* __restrict__ out, int n) {
    int i = blockIdx.x * blockDim.x + threadIdx.x;
    if (i < n) out[i] = f2bf(in[i]);
}

// ---------------------------------------------------------------------------
// C[M,N] = A[M,K] * B[N,K]^T   (both operands bf16 row-major; y = x @ W.T)
// Block: 256 thr = 8 waves (4x2), tile 128x128, BK=32. Each wave: 32x64.
// ---------------------------------------------------------------------------
template <bool OUT_BF16>
__global__ __launch_bounds__(256) void gemm_xwT(
    const unsigned short* __restrict__ A,
    const unsigned short* __restrict__ B,
    void* __restrict__ Cout, int M, int N, int K) {
    __shared__ unsigned short As[128][40];   // 32-wide + pad (80B rows, 16B aligned)
    __shared__ unsigned short Bs[128][40];

    const int tid  = threadIdx.x;
    const int lane = tid & 31;
    const int wid  = tid >> 5;
    const int wm   = wid >> 1;        // 0..3  (row band of 32)
    const int wn   = wid & 1;         // 0..1  (col band of 64)
    const int half = lane >> 4;
    const int l15  = lane & 15;
    const int row0 = blockIdx.y * 128;
    const int col0 = blockIdx.x * 128;

    const int lr = tid >> 1;          // 0..127 cooperative-load row
    const int lc = (tid & 1) * 16;    // 0 or 16

    const unsigned short* gA = A + (size_t)(row0 + lr) * K + lc;
    const unsigned short* gB = B + (size_t)(col0 + lr) * K + lc;

    v8f acc[2][4] = {};

    for (int kb = 0; kb < K; kb += 32) {
        // A tile 128x32, B tile 128x32: 16 bf16 per thread per tile.
        g2l_b128(&As[lr][lc],     gA + kb);
        g2l_b128(&As[lr][lc + 8], gA + kb + 8);
        g2l_b128(&Bs[lr][lc],     gB + kb);
        g2l_b128(&Bs[lr][lc + 8], gB + kb + 8);
        if (kb + 32 < K) {                       // hide L2 latency of next tile
            __builtin_prefetch(gA + kb + 32, 0, 1);
            __builtin_prefetch(gB + kb + 32, 0, 1);
        }
        async_join();
        __syncthreads();

        v16bf af[2], bf[4];
#pragma unroll
        for (int t = 0; t < 2; t++)
            af[t] = load_fragA(&As[wm * 32 + t * 16 + l15][0], half);
#pragma unroll
        for (int t = 0; t < 4; t++)
            bf[t] = load_fragB(&Bs[wn * 64 + t * 16 + l15][0], half);
#pragma unroll
        for (int tm = 0; tm < 2; tm++)
#pragma unroll
            for (int tn = 0; tn < 4; tn++)
                acc[tm][tn] = wmma_bf16(af[tm], bf[tn], acc[tm][tn]);
        __syncthreads();
    }

    // D layout: element r of v8f = row (r + 8*half), col = lane&15
#pragma unroll
    for (int tm = 0; tm < 2; tm++)
#pragma unroll
        for (int tn = 0; tn < 4; tn++) {
            FragF f; f.v = acc[tm][tn];
            int col = col0 + wn * 64 + tn * 16 + l15;
#pragma unroll
            for (int r = 0; r < 8; r++) {
                int row = row0 + wm * 32 + tm * 16 + r + 8 * half;
                if (OUT_BF16)
                    ((unsigned short*)Cout)[(size_t)row * N + col] = f2bf(f.f[r]);
                else
                    ((float*)Cout)[(size_t)row * N + col] = f.f[r];
            }
        }
}

// ---------------------------------------------------------------------------
// Flash attention, causal. qkv: [B*S][3072] bf16 (Q|K|V per head-major cols).
// Block = 128 thr = 4 waves; each wave owns 16 q-rows (q-tile 64). 32 keys/step.
// ---------------------------------------------------------------------------
__global__ __launch_bounds__(128) void flash_attn(
    const unsigned short* __restrict__ qkv,
    unsigned short* __restrict__ ctx) {
    const int S = 2048, E3 = 3072;
    const float SCALE = 0.125f;   // 1/sqrt(64)

    __shared__ unsigned short Ks[32][72];       // K tile rows (B^T for QK^T)
    __shared__ unsigned short Vt[64][40];       // V transposed (B^T for PV)
    __shared__ unsigned short Ps[4][16][40];    // wave-private P staging

    const int bh  = blockIdx.x;            // b*16 + h
    const int b   = bh >> 4, h = bh & 15;
    const int q0  = blockIdx.y * 64;
    const int tid = threadIdx.x;
    const int lane = tid & 31, w = tid >> 5;
    const int half = lane >> 4, l15 = lane & 15;

    // Q fragments for this wave's 16 rows, fixed for whole k-loop (d=0..63).
    Frag16 qf[2];
    {
        const unsigned short* qp =
            qkv + (size_t)(b * S + q0 + w * 16 + l15) * E3 + h * 64;
        qf[0].q[0] = *(const uint4*)(qp + half * 8);
        qf[0].q[1] = *(const uint4*)(qp + 16 + half * 8);
        qf[1].q[0] = *(const uint4*)(qp + 32 + half * 8);
        qf[1].q[1] = *(const uint4*)(qp + 48 + half * 8);
    }

    float mrow[8], lrow[8];
    v8f o[4];
#pragma unroll
    for (int r = 0; r < 8; r++) { mrow[r] = -INFINITY; lrow[r] = 0.0f; }
#pragma unroll
    for (int t = 0; t < 4; t++) o[t] = (v8f)0.0f;

    const int myq = q0 + w * 16;
    const int nkb = (q0 >> 5) + 2;   // keys needed: [0, q0+64)

    const int kk = tid >> 2;               // 0..31  cooperative-load key row
    const int db = (tid & 3) * 16;         // 0,16,32,48

    for (int kb = 0; kb < nkb; kb++) {
        const int kbase = kb * 32;
        // K tile: async global->LDS, 16 bf16 per thread
        {
            const unsigned short* kp =
                qkv + (size_t)(b * S + kbase + kk) * E3 + 1024 + h * 64 + db;
            g2l_b128(&Ks[kk][db],     kp);
            g2l_b128(&Ks[kk][db + 8], kp + 8);
        }
        // V tile: transposed into LDS (scatter -> manual)
        {
            const unsigned short* vp =
                qkv + (size_t)(b * S + kbase + kk) * E3 + 2048 + h * 64 + db;
            uint4 v0 = *(const uint4*)vp, v1 = *(const uint4*)(vp + 8);
            unsigned short tmp[16];
            *(uint4*)&tmp[0] = v0; *(uint4*)&tmp[8] = v1;
#pragma unroll
            for (int i = 0; i < 16; i++) Vt[db + i][kk] = tmp[i];
        }
        async_join();
        __syncthreads();

        // scores: 16q x 32k, accumulate over d=64 (two k=32 WMMA per n-tile)
        v8f s[2]; s[0] = (v8f)0.0f; s[1] = (v8f)0.0f;
#pragma unroll
        for (int t = 0; t < 2; t++) {
            const unsigned short* kr = &Ks[t * 16 + l15][0];
            Frag16 b0, b1;
            b0.q[0] = *(const uint4*)(kr + half * 8);
            b0.q[1] = *(const uint4*)(kr + 16 + half * 8);
            b1.q[0] = *(const uint4*)(kr + 32 + half * 8);
            b1.q[1] = *(const uint4*)(kr + 48 + half * 8);
            s[t] = wmma_bf16(qf[0].v, b0.v, s[t]);
            s[t] = wmma_bf16(qf[1].v, b1.v, s[t]);
        }

        // online softmax (row = r + 8*half lives across the lane's 16-lane half)
        FragF sf0, sf1; sf0.v = s[0]; sf1.v = s[1];
#pragma unroll
        for (int r = 0; r < 8; r++) {
            const int qi  = myq + r + 8 * half;
            const int k0i = kbase + l15;
            const int k1i = kbase + 16 + l15;
            float v0 = (k0i <= qi) ? sf0.f[r] * SCALE : -INFINITY;
            float v1 = (k1i <= qi) ? sf1.f[r] * SCALE : -INFINITY;

            float mx = fmaxf(v0, v1);
#pragma unroll
            for (int d = 8; d >= 1; d >>= 1) mx = fmaxf(mx, __shfl_xor(mx, d, 32));
            const float mnew = fmaxf(mrow[r], mx);
            const float corr = __expf(mrow[r] - mnew);   // first iter: exp(-inf)=0
            const float p0 = __expf(v0 - mnew);
            const float p1 = __expf(v1 - mnew);

            float rs = p0 + p1;
#pragma unroll
            for (int d = 8; d >= 1; d >>= 1) rs += __shfl_xor(rs, d, 32);
            lrow[r] = lrow[r] * corr + rs;
            mrow[r] = mnew;
#pragma unroll
            for (int t = 0; t < 4; t++) o[t][r] *= corr;

            // stage P (D-layout element -> row-major LDS), wave-private
            Ps[w][r + 8 * half][l15]      = f2bf(p0);
            Ps[w][r + 8 * half][16 + l15] = f2bf(p1);
        }

        // P (16x32) x V (32x64): reload P in A-fragment layout from LDS
        Frag16 pa;
        {
            const unsigned short* pr = &Ps[w][l15][0];
            pa.q[0] = *(const uint4*)(pr + half * 8);
            pa.q[1] = *(const uint4*)(pr + 16 + half * 8);
        }
#pragma unroll
        for (int t = 0; t < 4; t++) {
            const unsigned short* vr = &Vt[t * 16 + l15][0];
            Frag16 vb;
            vb.q[0] = *(const uint4*)(vr + half * 8);
            vb.q[1] = *(const uint4*)(vr + 16 + half * 8);
            o[t] = wmma_bf16(pa.v, vb.v, o[t]);
        }
        __syncthreads();
    }

    // normalize + write context [B*S][1024] bf16 (e = h*64 + d)
#pragma unroll
    for (int t = 0; t < 4; t++)
#pragma unroll
        for (int r = 0; r < 8; r++) {
            float val = o[t][r] / lrow[r];
            int row = q0 + w * 16 + r + 8 * half;
            int col = h * 64 + t * 16 + l15;
            ctx[(size_t)(b * S + row) * 1024 + col] = f2bf(val);
        }
}

// ---------------------------------------------------------------------------
extern "C" void kernel_launch(void* const* d_in, const int* in_sizes, int n_in,
                              void* d_out, int out_size, void* d_ws, size_t ws_size,
                              hipStream_t stream) {
    (void)in_sizes; (void)n_in; (void)out_size; (void)ws_size;
    const int B = 2, S = 2048, E = 1024;
    const int M = B * S;                     // 4096

    const float* x     = (const float*)d_in[0];
    const float* w_qkv = (const float*)d_in[1];   // [3E, E]
    const float* w_out = (const float*)d_in[2];   // [E, E]
    float* out = (float*)d_out;

    unsigned short* xb    = (unsigned short*)d_ws;               // [M][E]
    unsigned short* wqkvb = xb    + (size_t)M * E;               // [3E][E]
    unsigned short* woutb = wqkvb + (size_t)3 * E * E;           // [E][E]
    unsigned short* qkvb  = woutb + (size_t)E * E;               // [M][3E]
    unsigned short* ctxb  = qkvb  + (size_t)M * 3 * E;           // [M][E]

    const int n_x = M * E, n_wq = 3 * E * E, n_wo = E * E;
    cvt_f32_bf16<<<(n_x  + 255) / 256, 256, 0, stream>>>(x,     xb,    n_x);
    cvt_f32_bf16<<<(n_wq + 255) / 256, 256, 0, stream>>>(w_qkv, wqkvb, n_wq);
    cvt_f32_bf16<<<(n_wo + 255) / 256, 256, 0, stream>>>(w_out, woutb, n_wo);

    // qkv = x @ W_qkv^T  -> bf16 [4096][3072]
    gemm_xwT<true><<<dim3(3 * E / 128, M / 128), 256, 0, stream>>>(
        xb, wqkvb, qkvb, M, 3 * E, E);

    // flash attention -> ctx bf16 [4096][1024]
    flash_attn<<<dim3(B * 16, S / 64), 128, 0, stream>>>(qkvb, ctxb);

    // out = ctx @ W_out^T -> f32 [4096][1024]
    gemm_xwT<false><<<dim3(E / 128, M / 128), 256, 0, stream>>>(
        ctxb, woutb, out, M, E, E);
}
